// Block_4569845203333
// MI455X (gfx1250) — matrix operand order repaired and checked by reading, not compiled, not run on hardware
//
#include <hip/hip_runtime.h>

// ---------------------------------------------------------------------------
// Types for CDNA5 WMMA (wave32): bf16 16x16x32, fp32 accumulate.
// ---------------------------------------------------------------------------
typedef __attribute__((ext_vector_type(16))) __bf16       v16bf;
typedef __attribute__((ext_vector_type(4)))  __bf16       v4bf;
typedef __attribute__((ext_vector_type(8)))  float        v8f;
typedef __attribute__((ext_vector_type(4)))  unsigned int u32x4;

union Frag16 {              // 16 bf16 elements = 8 VGPRs
    v16bf v;
    u32x4 u[2];
};

static __device__ __forceinline__ v8f wmma_bf16(const v16bf& a, const v16bf& b, const v8f& c) {
    return __builtin_amdgcn_wmma_f32_16x16x32_bf16(
        /*neg_a=*/false, a, /*neg_b=*/false, b,
        /*c_mod=*/(short)0, c, /*reuse_a=*/false, /*reuse_b=*/false);
}

// ---------------------------------------------------------------------------
// 16-lane butterfly reductions via v_permlane16_b32 (VALU, co-executes with
// WMMA; keeps the DS pipe free for the P-fragment staging).
// sel nibble i = source lane (i ^ k) within each row of 16.
// ---------------------------------------------------------------------------
static __device__ __forceinline__ float permlane16_f(float v, int s0, int s1) {
    int x = __float_as_int(v);
    x = __builtin_amdgcn_permlane16(x, x, s0, s1, false, false);
    return __int_as_float(x);
}
static __device__ __forceinline__ float rowmax16(float v) {
    v = fmaxf(v, permlane16_f(v, 0x67452301, 0xEFCDAB89));  // xor 1
    v = fmaxf(v, permlane16_f(v, 0x54761032, 0xDCFE98BA));  // xor 2
    v = fmaxf(v, permlane16_f(v, 0x32107654, 0xBA98FEDC));  // xor 4
    v = fmaxf(v, permlane16_f(v, 0xFEDCBA98, 0x76543210));  // xor 8
    return v;
}
static __device__ __forceinline__ float rowsum16(float v) {
    v += permlane16_f(v, 0x67452301, 0xEFCDAB89);           // xor 1
    v += permlane16_f(v, 0x54761032, 0xDCFE98BA);           // xor 2
    v += permlane16_f(v, 0x32107654, 0xBA98FEDC);           // xor 4
    v += permlane16_f(v, 0xFEDCBA98, 0x76543210);           // xor 8
    return v;
}

// ---------------------------------------------------------------------------
// fp32 -> bf16 elementwise conversion (weights)
// ---------------------------------------------------------------------------
__global__ __launch_bounds__(256)
void f2bf_kernel(const float* __restrict__ src, __bf16* __restrict__ dst, int n) {
    int i = blockIdx.x * 256 + threadIdx.x;
    if (i < n) dst[i] = (__bf16)src[i];
}

// ---------------------------------------------------------------------------
// LayerNorm over last dim D (multiple of 1024); one row per block, bf16 out.
// ---------------------------------------------------------------------------
__global__ __launch_bounds__(256)
void ln_kernel(const float* __restrict__ x, const float* __restrict__ g,
               const float* __restrict__ bta, __bf16* __restrict__ out, int D) {
    __shared__ float s1[256];
    __shared__ float s2[256];
    const int tid = threadIdx.x;
    const float* xr = x + (size_t)blockIdx.x * D;

    float sum = 0.f, sq = 0.f;
    for (int i = tid * 4; i < D; i += 256 * 4) {
        float4 v = *(const float4*)(const void*)(xr + i);
        sum += v.x + v.y + v.z + v.w;
        sq  += v.x * v.x + v.y * v.y + v.z * v.z + v.w * v.w;
    }
    s1[tid] = sum; s2[tid] = sq;
    __syncthreads();
    for (int off = 128; off > 0; off >>= 1) {
        if (tid < off) { s1[tid] += s1[tid + off]; s2[tid] += s2[tid + off]; }
        __syncthreads();
    }
    const float mu  = s1[0] / (float)D;
    const float var = s2[0] / (float)D - mu * mu;
    const float rs  = rsqrtf(var + 1e-5f);

    __bf16* orow = out + (size_t)blockIdx.x * D;
    for (int i = tid * 4; i < D; i += 256 * 4) {
        float4 v = *(const float4*)(const void*)(xr + i);
        float4 gg = *(const float4*)(const void*)(g + i);
        float4 bb = *(const float4*)(const void*)(bta + i);
        v4bf o;
        o[0] = (__bf16)((v.x - mu) * rs * gg.x + bb.x);
        o[1] = (__bf16)((v.y - mu) * rs * gg.y + bb.y);
        o[2] = (__bf16)((v.z - mu) * rs * gg.z + bb.z);
        o[3] = (__bf16)((v.w - mu) * rs * gg.w + bb.w);
        *(v4bf*)(void*)(orow + i) = o;
    }
}

// ---------------------------------------------------------------------------
// GEMM: out[M,Nn] = A[M,K] @ B[Nn,K]^T  (A,B bf16 row-major, out fp32)
// EPI: 0 = none, 1 = +bias then GELU(erf), 2 = +bias then +residual
// Block: 256 threads = 8 waves; wave tile 32x64; WG tile 128x128.
// K-loop unrolled x2 with register double-buffering (K % 64 == 0).
// ---------------------------------------------------------------------------
template<int EPI>
__global__ __launch_bounds__(256)
void gemm_bf16_kernel(const __bf16* __restrict__ A, const __bf16* __restrict__ Bm,
                      const float* __restrict__ bias, const float* __restrict__ res,
                      float* __restrict__ out, int M, int Nn, int K) {
    const int lane = threadIdx.x & 31;
    const int wid  = threadIdx.x >> 5;
    const int half = lane >> 4;      // 0: lanes 0-15, 1: lanes 16-31
    const int l16  = lane & 15;
    const int wm   = wid & 3;        // 4 waves along M
    const int wn   = wid >> 2;       // 2 waves along N

    const int mBase = blockIdx.y * 128 + wm * 32;
    const int nBase = blockIdx.x * 128 + wn * 64;

    // Per-lane base pointers (A and B: elems 0-7 at K=half*8, 8-15 at K=16+half*8).
    const __bf16* aPtr[2];
    const __bf16* bPtr[4];
    #pragma unroll
    for (int ma = 0; ma < 2; ++ma)
        aPtr[ma] = A + (size_t)(mBase + ma * 16 + l16) * K + half * 8;
    #pragma unroll
    for (int nf = 0; nf < 4; ++nf)
        bPtr[nf] = Bm + (size_t)(nBase + nf * 16 + l16) * K + half * 8;

    auto loadA = [&](Frag16* a, int k) {
        #pragma unroll
        for (int ma = 0; ma < 2; ++ma) {
            a[ma].u[0] = *(const u32x4*)(const void*)(aPtr[ma] + k);
            a[ma].u[1] = *(const u32x4*)(const void*)(aPtr[ma] + k + 16);
        }
    };
    auto loadB = [&](Frag16* b, int k) {
        #pragma unroll
        for (int nf = 0; nf < 4; ++nf) {
            b[nf].u[0] = *(const u32x4*)(const void*)(bPtr[nf] + k);
            b[nf].u[1] = *(const u32x4*)(const void*)(bPtr[nf] + k + 16);
        }
    };

    v8f acc[2][4];
    #pragma unroll
    for (int ma = 0; ma < 2; ++ma)
        #pragma unroll
        for (int nf = 0; nf < 4; ++nf)
            acc[ma][nf] = (v8f){};

    Frag16 a0[2], b0[4], a1[2], b1[4];
    loadA(a0, 0);
    loadB(b0, 0);

    for (int k = 0; k < K; k += 64) {
        loadA(a1, k + 32);
        loadB(b1, k + 32);
        #pragma unroll
        for (int ma = 0; ma < 2; ++ma)
            #pragma unroll
            for (int nf = 0; nf < 4; ++nf)
                acc[ma][nf] = wmma_bf16(a0[ma].v, b0[nf].v, acc[ma][nf]);
        if (k + 64 < K) {
            loadA(a0, k + 64);
            loadB(b0, k + 64);
        }
        #pragma unroll
        for (int ma = 0; ma < 2; ++ma)
            #pragma unroll
            for (int nf = 0; nf < 4; ++nf)
                acc[ma][nf] = wmma_bf16(a1[ma].v, b1[nf].v, acc[ma][nf]);
    }

    // Epilogue + store (C/D layout: lane l, VGPR v -> row = v + 8*half, col = l16)
    #pragma unroll
    for (int nf = 0; nf < 4; ++nf) {
        const int col = nBase + nf * 16 + l16;
        float bv = 0.f;
        if (EPI != 0) bv = bias[col];
        #pragma unroll
        for (int ma = 0; ma < 2; ++ma) {
            #pragma unroll
            for (int v = 0; v < 8; ++v) {
                const int row = mBase + ma * 16 + v + half * 8;
                const size_t idx = (size_t)row * Nn + col;
                float val = acc[ma][nf][v];
                if (EPI == 1) {
                    val += bv;
                    val = 0.5f * val * (1.0f + erff(val * 0.70710678118654752f));
                } else if (EPI == 2) {
                    val += bv + res[idx];
                }
                out[idx] = val;
            }
        }
    }
}

// ---------------------------------------------------------------------------
// Rearrange qkv fp32 [B,N,3,H,Dh] -> q/k [B,H,N,Dh] bf16 and vT [B,H,Dh,N] bf16
// ---------------------------------------------------------------------------
__global__ __launch_bounds__(256)
void qkv_rearrange_kernel(const float* __restrict__ qkvf, __bf16* __restrict__ qd,
                          __bf16* __restrict__ kd, __bf16* __restrict__ vtd) {
    const int idx   = blockIdx.x * 256 + threadIdx.x;   // < 4096*1024
    const int c     = idx & 1023;
    const int token = idx >> 10;
    const int b     = token >> 10;
    const int n     = token & 1023;
    const int h     = c >> 6;
    const int d     = c & 63;
    const float* base = qkvf + (size_t)token * 3072;
    const size_t bh = (size_t)(b * 16 + h);
    qd[(bh * 1024 + n) * 64 + d]  = (__bf16)base[c];
    kd[(bh * 1024 + n) * 64 + d]  = (__bf16)base[1024 + c];
    vtd[(bh * 64 + d) * 1024 + n] = (__bf16)base[2048 + c];
}

// ---------------------------------------------------------------------------
// Flash-style attention, one (b,h) per blockIdx.x, 128 query rows per block,
// 16 query rows per wave. Dh = 64, N = 1024, scale = 1/8.
// S=QK^T via WMMA; online softmax with v_permlane16 row reductions; P relaid
// C-frag -> A-frag through per-wave LDS; O += P @ V via WMMA (V transposed).
// Output written merged: O[b, n, h*64 + d] bf16.
// ---------------------------------------------------------------------------
__global__ __launch_bounds__(256)
void attn_kernel(const __bf16* __restrict__ q, const __bf16* __restrict__ kmat,
                 const __bf16* __restrict__ vT, __bf16* __restrict__ Obf) {
    __shared__ __bf16 pbuf[8][16 * 32];   // per-wave P staging (1 KB each)

    const int lane = threadIdx.x & 31;
    const int wid  = threadIdx.x >> 5;
    const int half = lane >> 4;
    const int l16  = lane & 15;

    const int bh = blockIdx.x;            // 0..63
    const int b  = bh >> 4;
    const int h  = bh & 15;
    const int i0 = blockIdx.y * 128 + wid * 16;

    const size_t qkBase = (size_t)bh * 1024 * 64;
    const size_t vtBase = (size_t)bh * 64 * 1024;
    const float scale = 0.125f;           // 1/sqrt(64)

    // Q A-fragments for the two K(=d) steps of 32
    Frag16 qa[2];
    #pragma unroll
    for (int t = 0; t < 2; ++t) {
        const __bf16* p = q + qkBase + (size_t)(i0 + l16) * 64 + t * 32 + half * 8;
        qa[t].u[0] = *(const u32x4*)(const void*)p;
        qa[t].u[1] = *(const u32x4*)(const void*)(p + 16);
    }

    float mrow[8], srow[8];
    #pragma unroll
    for (int v = 0; v < 8; ++v) { mrow[v] = -3.0e38f; srow[v] = 0.f; }
    v8f o[4];
    #pragma unroll
    for (int c = 0; c < 4; ++c) o[c] = (v8f){};

    for (int j0 = 0; j0 < 1024; j0 += 32) {
        // ---- S = Q K^T for 32 keys (two 16x16 tiles, 4 WMMAs) ----
        v8f s[2];
        #pragma unroll
        for (int jt = 0; jt < 2; ++jt) {
            s[jt] = (v8f){};
            #pragma unroll
            for (int kk = 0; kk < 2; ++kk) {
                Frag16 bf;
                const __bf16* p = kmat + qkBase +
                    (size_t)(j0 + jt * 16 + l16) * 64 + kk * 32 + half * 8;
                bf.u[0] = *(const u32x4*)(const void*)p;
                bf.u[1] = *(const u32x4*)(const void*)(p + 16);
                s[jt] = wmma_bf16(qa[kk].v, bf.v, s[jt]);
            }
        }

        // ---- prefetch V^T B-fragments (independent of softmax) ----
        Frag16 vfrag[4];
        #pragma unroll
        for (int c = 0; c < 4; ++c) {
            const __bf16* p = vT + vtBase + (size_t)(c * 16 + l16) * 1024 + j0 + half * 8;
            vfrag[c].u[0] = *(const u32x4*)(const void*)p;
            vfrag[c].u[1] = *(const u32x4*)(const void*)(p + 16);
        }

        // ---- online softmax update (row = v + 8*half across 16-lane group) ----
        #pragma unroll
        for (int v = 0; v < 8; ++v) {
            float sa = s[0][v] * scale;
            float sb = s[1][v] * scale;
            float tmax = rowmax16(fmaxf(sa, sb));
            const float mn    = fmaxf(mrow[v], tmax);
            const float alpha = __expf(mrow[v] - mn);
            const float pa    = __expf(sa - mn);
            const float pb    = __expf(sb - mn);
            const float rsum  = rowsum16(pa + pb);
            srow[v] = srow[v] * alpha + rsum;
            mrow[v] = mn;
            #pragma unroll
            for (int c = 0; c < 4; ++c) o[c][v] *= alpha;

            const int r = v + half * 8;
            pbuf[wid][r * 32 + l16]      = (__bf16)pa;
            pbuf[wid][r * 32 + 16 + l16] = (__bf16)pb;
        }

        // ---- reload P as an A-fragment (16 x 32, K = key index) ----
        Frag16 pf;
        {
            const __bf16* lp = &pbuf[wid][l16 * 32];
            unsigned int* pw = (unsigned int*)&pf;
            #pragma unroll
            for (int e = 0; e < 4; ++e)
                pw[e] = *(const unsigned int*)(const void*)(lp + half * 8 + 2 * e);
            #pragma unroll
            for (int e = 0; e < 4; ++e)
                pw[4 + e] = *(const unsigned int*)(const void*)(lp + 16 + half * 8 + 2 * e);
        }

        // ---- O += P @ V (4 WMMAs over the four d-chunks of 16) ----
        #pragma unroll
        for (int c = 0; c < 4; ++c)
            o[c] = wmma_bf16(pf.v, vfrag[c].v, o[c]);
    }

    // ---- normalize and write merged-head output (bf16) ----
    float invr[8];
    #pragma unroll
    for (int v = 0; v < 8; ++v) invr[v] = 1.0f / srow[v];
    #pragma unroll
    for (int c = 0; c < 4; ++c) {
        const int col = h * 64 + c * 16 + l16;
        #pragma unroll
        for (int v = 0; v < 8; ++v) {
            const int row = i0 + v + half * 8;
            Obf[(size_t)(b * 1024 + row) * 1024 + col] = (__bf16)(o[c][v] * invr[v]);
        }
    }
}

// ---------------------------------------------------------------------------
// Host-side orchestration
// ---------------------------------------------------------------------------
extern "C" void kernel_launch(void* const* d_in, const int* in_sizes, int n_in,
                              void* d_out, int out_size, void* d_ws, size_t ws_size,
                              hipStream_t stream) {
    (void)in_sizes; (void)n_in; (void)out_size; (void)ws_size;
    const float* x      = (const float*)d_in[0];
    const float* qkv_w  = (const float*)d_in[1];
    const float* proj_w = (const float*)d_in[2];
    const float* proj_b = (const float*)d_in[3];
    const float* fc1_w  = (const float*)d_in[4];
    const float* fc1_b  = (const float*)d_in[5];
    const float* fc2_w  = (const float*)d_in[6];
    const float* fc2_b  = (const float*)d_in[7];
    const float* ln1_g  = (const float*)d_in[8];
    const float* ln1_b  = (const float*)d_in[9];
    const float* ln2_g  = (const float*)d_in[10];
    const float* ln2_b  = (const float*)d_in[11];
    const float* lnh_g  = (const float*)d_in[12];
    const float* lnh_b  = (const float*)d_in[13];
    float* out = (float*)d_out;

    // Workspace layout (bump allocated, with region reuse):
    char* ws = (char*)d_ws;
    size_t off = 0;
    __bf16* wqkv  = (__bf16*)(ws + off); off += (size_t)3072 * 1024 * 2;
    __bf16* wproj = (__bf16*)(ws + off); off += (size_t)1024 * 1024 * 2;
    __bf16* wfc1  = (__bf16*)(ws + off); off += (size_t)4096 * 1024 * 2;
    __bf16* wfc2  = (__bf16*)(ws + off); off += (size_t)1024 * 4096 * 2;
    __bf16* hbf   = (__bf16*)(ws + off); off += (size_t)4096 * 1024 * 2;   // h1, reused as h2
    float*  Rf    = (float*)(ws + off);  off += (size_t)4096 * 4096 * 4;   // qkv fp32 (48MB), reused as u fp32 (64MB)
    const size_t headBytes = (size_t)64 * 1024 * 64 * 2;                   // 8 MB each
    __bf16* qbf  = (__bf16*)(ws + off);
    __bf16* kbf  = (__bf16*)(ws + off + headBytes);
    __bf16* vtbf = (__bf16*)(ws + off + 2 * headBytes);
    __bf16* obf  = (__bf16*)(ws + off + 3 * headBytes);
    __bf16* ubf  = qbf;                                                    // 32 MB region reused for u bf16
    off += 4 * headBytes;
    float* x1 = (float*)(ws + off); off += (size_t)4096 * 1024 * 4;

    // 1) Weight conversions fp32 -> bf16
    f2bf_kernel<<<(3072 * 1024 + 255) / 256, 256, 0, stream>>>(qkv_w, wqkv, 3072 * 1024);
    f2bf_kernel<<<(1024 * 1024 + 255) / 256, 256, 0, stream>>>(proj_w, wproj, 1024 * 1024);
    f2bf_kernel<<<(4096 * 1024 + 255) / 256, 256, 0, stream>>>(fc1_w, wfc1, 4096 * 1024);
    f2bf_kernel<<<(1024 * 4096 + 255) / 256, 256, 0, stream>>>(fc2_w, wfc2, 1024 * 4096);

    // 2) LN1: x -> h (bf16)
    ln_kernel<<<4096, 256, 0, stream>>>(x, ln1_g, ln1_b, hbf, 1024);

    // 3) QKV GEMM: qkv = h @ qkv_w^T   [4096 x 3072]
    gemm_bf16_kernel<0><<<dim3(3072 / 128, 4096 / 128), 256, 0, stream>>>(
        hbf, wqkv, nullptr, nullptr, Rf, 4096, 3072, 1024);

    // 4) Rearrange to q/k [B,H,N,Dh] and vT [B,H,Dh,N]
    qkv_rearrange_kernel<<<(4096 * 1024) / 256, 256, 0, stream>>>(Rf, qbf, kbf, vtbf);

    // 5) Attention -> merged O (bf16)
    attn_kernel<<<dim3(64, 8), 256, 0, stream>>>(qbf, kbf, vtbf, obf);

    // 6) Projection + residual: x1 = x + O @ proj_w^T + proj_b
    gemm_bf16_kernel<2><<<dim3(1024 / 128, 4096 / 128), 256, 0, stream>>>(
        obf, wproj, proj_b, x, x1, 4096, 1024, 1024);

    // 7) LN2: x1 -> h2 (bf16, reuse hbf)
    ln_kernel<<<4096, 256, 0, stream>>>(x1, ln2_g, ln2_b, hbf, 1024);

    // 8) FC1 + GELU(erf): u = gelu(h2 @ fc1_w^T + fc1_b)  [4096 x 4096] fp32
    gemm_bf16_kernel<1><<<dim3(4096 / 128, 4096 / 128), 256, 0, stream>>>(
        hbf, wfc1, fc1_b, nullptr, Rf, 4096, 4096, 1024);

    // 9) LN over hidden: u -> u_bf (bf16)
    ln_kernel<<<4096, 256, 0, stream>>>(Rf, lnh_g, lnh_b, ubf, 4096);

    // 10) FC2 + residual: out = x1 + u_bf @ fc2_w^T + fc2_b
    gemm_bf16_kernel<2><<<dim3(1024 / 128, 4096 / 128), 256, 0, stream>>>(
        ubf, wfc2, fc2_b, x1, out, 4096, 1024, 4096);
}
